// VarianceAdaptor_softplus1_21655225106572
// MI455X (gfx1250) — matrix operand order repaired and checked by reading, not compiled, not run on hardware
//
#include <hip/hip_runtime.h>
#include <cstdint>

// ---------------------------------------------------------------------------
// VarianceAdaptor for MI455X (gfx1250, wave32)
//   B=64, L=200, H=256, MAXLEN=2000
//   Part 1: fused duration predictor (WMMA f32<-f16 GEMMs + LN + softplus)
//   Part 2: per-batch cumsum of durations
//   Part 3: length-regulate gather (bandwidth-bound, NT B128 stores)
// ---------------------------------------------------------------------------

typedef __attribute__((ext_vector_type(16))) _Float16 v16h;
typedef __attribute__((ext_vector_type(8)))  float    v8f;
typedef __attribute__((ext_vector_type(4)))  float    v4f;

#define BATCH   64
#define SEQL    200
#define HID     256
#define MAXLEN  2000
#define ROWS    (BATCH * SEQL)        // 12800
#define NBLK_DP (ROWS / 16)           // 800 blocks of 16 rows

__device__ __forceinline__ float softplusf(float z) {
    if (z > 20.0f)  return z;
    if (z < -20.0f) return expf(z);
    return log1pf(expf(z));
}

// ---------------------------------------------------------------------------
// Fused duration predictor.
// Block = 256 threads = 8 waves; owns 16 consecutive rows of (ROWS x HID).
// Each wave computes two 16x16 output tiles (N covered: 8 waves * 32 = 256).
// ---------------------------------------------------------------------------
__global__ __launch_bounds__(256)
void dp_kernel(const float* __restrict__ y,            // [ROWS, HID]
               const unsigned char* __restrict__ src_mask,  // [ROWS]
               const float* __restrict__ W1, const float* __restrict__ b1,
               const float* __restrict__ g1, const float* __restrict__ be1,
               const float* __restrict__ W2, const float* __restrict__ b2,
               const float* __restrict__ g2, const float* __restrict__ be2,
               const float* __restrict__ Wc, const float* __restrict__ bc,
               float* __restrict__ logdur)              // [ROWS]
{
    __shared__ float    tile[16 * HID];     // fp32 activations (16 KB)
    __shared__ _Float16 actf16[16 * HID];   // f16 A-matrix for WMMA (8 KB)
    __shared__ float    red1[256];
    __shared__ float    red2[256];
    __shared__ float    mean_s[16];
    __shared__ float    rstd_s[16];

    const int tid  = threadIdx.x;
    const int lane = tid & 31;
    const int wave = tid >> 5;
    const int lrow = lane & 15;   // M (for A) or N (for B) within tile
    const int kgrp = lane >> 4;   // 0 or 1
    const int r0   = blockIdx.x * 16;

    // Stage the 16 input rows as f16 (layer-1 A matrix)
    for (int i = tid; i < 16 * HID; i += 256) {
        actf16[i] = (_Float16)y[(size_t)(r0 + (i >> 8)) * HID + (i & 255)];
    }
    __syncthreads();

    const int nt0 = wave * 2;
    const int nt1 = wave * 2 + 1;
    const int n0  = nt0 * 16 + lrow;      // global N of this lane, tile 0
    const int n1  = nt1 * 16 + lrow;      // global N of this lane, tile 1

    #pragma unroll 1
    for (int layer = 0; layer < 2; ++layer) {
        const float* W  = layer ? W2  : W1;
        const float* bb = layer ? b2  : b1;
        const float* gg = layer ? g2  : g1;
        const float* be = layer ? be2 : be1;

        v8f acc0 = {};
        v8f acc1 = {};

        // K loop: 256 / 32 = 8 WMMA steps per tile
        #pragma unroll 1
        for (int kc = 0; kc < 8; ++kc) {
            const int k0 = kc * 32;
            v16h a, bf0, bf1;
            #pragma unroll
            for (int e = 0; e < 16; ++e) {
                // A-matrix 16x32 f16 layout (ISA 7.12.2):
                //   VGPR0..3 -> K = e + kgrp*8 ; VGPR4..7 -> +8 more
                const int ka = k0 + e + kgrp * 8 + ((e >= 8) ? 8 : 0);
                a[e] = actf16[lrow * HID + ka];
                // B-matrix 32x16 f16 layout: same K order as A (K = e + kgrp*8, +8 for e>=8)
                const int kb = k0 + e + kgrp * 8 + ((e >= 8) ? 8 : 0);
                // out[m,f] = sum_h act[m,h] * W[f,h]  =>  B[k][n] = W[n][k]
                bf0[e] = (_Float16)W[(size_t)n0 * HID + kb];
                bf1[e] = (_Float16)W[(size_t)n1 * HID + kb];
            }
            acc0 = __builtin_amdgcn_wmma_f32_16x16x32_f16(
                       false, a, false, bf0, (short)0, acc0, false, false);
            acc1 = __builtin_amdgcn_wmma_f32_16x16x32_f16(
                       false, a, false, bf1, (short)0, acc1, false, false);
        }

        // D fragment -> LDS with bias + ReLU.
        // C/D layout: VGPR i holds row (i + kgrp*8), col = lane&15 of the tile.
        const float bias0 = bb[n0];
        const float bias1 = bb[n1];
        #pragma unroll
        for (int i = 0; i < 8; ++i) {
            const int r = i + kgrp * 8;
            float v0 = acc0[i] + bias0; v0 = v0 > 0.0f ? v0 : 0.0f;
            float v1 = acc1[i] + bias1; v1 = v1 > 0.0f ? v1 : 0.0f;
            tile[r * HID + n0] = v0;
            tile[r * HID + n1] = v1;
        }
        __syncthreads();

        // LayerNorm over HID per row: thread t -> row tid>>4, 16-elem segment tid&15
        const int rr  = tid >> 4;
        const int seg = tid & 15;
        float s1 = 0.0f, s2 = 0.0f;
        #pragma unroll
        for (int j = 0; j < 16; ++j) {
            const float v = tile[rr * HID + seg * 16 + j];
            s1 += v; s2 += v * v;
        }
        red1[tid] = s1;
        red2[tid] = s2;
        __syncthreads();
        if (tid < 16) {
            float a1 = 0.0f, a2 = 0.0f;
            #pragma unroll
            for (int j = 0; j < 16; ++j) { a1 += red1[tid * 16 + j]; a2 += red2[tid * 16 + j]; }
            const float m = a1 * (1.0f / HID);
            const float v = a2 * (1.0f / HID) - m * m;
            mean_s[tid] = m;
            rstd_s[tid] = rsqrtf(v + 1e-5f);
        }
        __syncthreads();
        // Normalize; keep fp32 in tile (for projection) and f16 (next GEMM's A)
        for (int i = tid; i < 16 * HID; i += 256) {
            const int r = i >> 8;
            const int c = i & 255;
            const float v = (tile[i] - mean_s[r]) * rstd_s[r] * gg[c] + be[c];
            tile[i]   = v;
            actf16[i] = (_Float16)v;
        }
        __syncthreads();
    }

    // Projection to 1 channel + softplus + src_mask
    {
        const int rr  = tid >> 4;
        const int seg = tid & 15;
        float p = 0.0f;
        #pragma unroll
        for (int j = 0; j < 16; ++j)
            p += tile[rr * HID + seg * 16 + j] * Wc[seg * 16 + j];
        red1[tid] = p;
        __syncthreads();
        if (tid < 16) {
            float z = bc[0];
            #pragma unroll
            for (int j = 0; j < 16; ++j) z += red1[tid * 16 + j];
            const int row = r0 + tid;
            logdur[row] = src_mask[row] ? 0.0f : softplusf(z);
        }
    }
}

// ---------------------------------------------------------------------------
// Per-batch cumulative sum of durations; also emits duration_rounded (float)
// and mel_len (int into ws, float into d_out).
// ---------------------------------------------------------------------------
__global__ __launch_bounds__(64)
void cumsum_kernel(const int* __restrict__ dur,      // [BATCH, SEQL]
                   int* __restrict__ cum,            // ws [BATCH, SEQL]
                   int* __restrict__ mlen,           // ws [BATCH]
                   float* __restrict__ dur_out,      // [BATCH, SEQL]
                   float* __restrict__ mellen_out)   // [BATCH]
{
    const int b = threadIdx.x;
    if (b >= BATCH) return;
    int acc = 0;
    for (int l = 0; l < SEQL; ++l) {
        const int d = dur[b * SEQL + l];
        acc += d;
        cum[b * SEQL + l] = acc;
        dur_out[b * SEQL + l] = (float)d;
    }
    mlen[b]       = acc;
    mellen_out[b] = (float)acc;
}

// ---------------------------------------------------------------------------
// Length regulate: one wave per output row (256 floats = 2x v4f per lane).
// Redundant per-lane binary search over L2-cached cumsum, NT B128 stores for
// the 131 MB write-once output stream.
// ---------------------------------------------------------------------------
__global__ __launch_bounds__(256)
void gather_kernel(const float* __restrict__ x,              // [BATCH, SEQL, HID]
                   const int* __restrict__ cum,              // ws
                   const int* __restrict__ mlen,             // ws
                   const unsigned char* __restrict__ mel_mask_in,  // [BATCH, MAXLEN]
                   float* __restrict__ xout,                 // [BATCH, MAXLEN, HID]
                   float* __restrict__ melmask_out)          // [BATCH, MAXLEN]
{
    const int rowid = blockIdx.x * 8 + (threadIdx.x >> 5);
    const int lane  = threadIdx.x & 31;
    const int b = rowid / MAXLEN;
    const int t = rowid % MAXLEN;

    v4f out0 = {0.f, 0.f, 0.f, 0.f};
    v4f out1 = {0.f, 0.f, 0.f, 0.f};

    if (t < mlen[b]) {
        // searchsorted(cum, t, 'right'): first idx with cum[idx] > t
        const int* c = cum + b * SEQL;
        int lo = 0, hi = SEQL;
        while (lo < hi) {
            const int mid = (lo + hi) >> 1;
            if (c[mid] <= t) lo = mid + 1; else hi = mid;
        }
        const int idx = lo < (SEQL - 1) ? lo : (SEQL - 1);
        const v4f* src = (const v4f*)(x + ((size_t)b * SEQL + idx) * HID);
        out0 = src[lane];
        out1 = src[lane + 32];
    }

    v4f* dst = (v4f*)(xout + ((size_t)b * MAXLEN + t) * HID);
    __builtin_nontemporal_store(out0, dst + lane);
    __builtin_nontemporal_store(out1, dst + lane + 32);

    if (lane == 0) {
        melmask_out[rowid] = mel_mask_in[rowid] ? 1.0f : 0.0f;
    }
}

// ---------------------------------------------------------------------------
extern "C" void kernel_launch(void* const* d_in, const int* in_sizes, int n_in,
                              void* d_out, int out_size, void* d_ws, size_t ws_size,
                              hipStream_t stream) {
    (void)in_sizes; (void)n_in; (void)out_size; (void)ws_size;

    const float*         x        = (const float*)d_in[0];
    const float*         y_lip    = (const float*)d_in[1];
    const unsigned char* src_mask = (const unsigned char*)d_in[2];
    const unsigned char* mel_mask = (const unsigned char*)d_in[3];
    const int*           dur      = (const int*)d_in[4];
    // d_in[5] = max_len scalar (2000), compile-time constant here
    const float* W1 = (const float*)d_in[6];
    const float* b1 = (const float*)d_in[7];
    const float* g1 = (const float*)d_in[8];
    const float* be1= (const float*)d_in[9];
    const float* W2 = (const float*)d_in[10];
    const float* b2 = (const float*)d_in[11];
    const float* g2 = (const float*)d_in[12];
    const float* be2= (const float*)d_in[13];
    const float* Wc = (const float*)d_in[14];
    const float* bc = (const float*)d_in[15];

    // Output layout (tuple flattened, all f32):
    float* out        = (float*)d_out;
    float* xout       = out;                                   // 32,768,000
    float* logdur     = xout + (size_t)BATCH * MAXLEN * HID;   // 12,800
    float* dur_out    = logdur + ROWS;                         // 12,800
    float* mellen_out = dur_out + ROWS;                        // 64
    float* melmask_out= mellen_out + BATCH;                    // 128,000

    // Workspace: cumsum + mel_len
    int* cum  = (int*)d_ws;          // BATCH*SEQL ints
    int* mlen = cum + ROWS;          // BATCH ints

    cumsum_kernel<<<1, 64, 0, stream>>>(dur, cum, mlen, dur_out, mellen_out);

    dp_kernel<<<NBLK_DP, 256, 0, stream>>>(y_lip, src_mask,
                                           W1, b1, g1, be1,
                                           W2, b2, g2, be2,
                                           Wc, bc, logdur);

    gather_kernel<<<(BATCH * MAXLEN) / 8, 256, 0, stream>>>(
        x, cum, mlen, mel_mask, xout, melmask_out);
}
